// SelfAttentionLayer_5463198401142
// MI455X (gfx1250) — compile-verified, matrix operand order checked by reading, not run on hardware
//
#include <hip/hip_runtime.h>
#include <hip/hip_bf16.h>
#include <math.h>

typedef __attribute__((ext_vector_type(16))) __bf16 v16bf;
typedef __attribute__((ext_vector_type(8)))  __bf16 v8bf;
typedef __attribute__((ext_vector_type(8)))  float  v8f;

#define WMMA_BF16(a, b, c) \
  __builtin_amdgcn_wmma_f32_16x16x32_bf16(false, (a), false, (b), (short)0, (c), false, false)

// B=16, L=1024, H=512, NH=8, DH=64, M=2048
#define CB  16
#define CL  1024
#define CH  512
#define CNH 8
#define CDH 64
#define CM  2048
#define NTOK (CB * CL)   // 16384

// ---- fragment loader: contraction dim contiguous, chunks at +0 and +16 elems ----
__device__ __forceinline__ v16bf load_frag(const __bf16* p) {
  union { v16bf v; v8bf h[2]; } u;
  u.h[0] = *(const v8bf*)(p);
  u.h[1] = *(const v8bf*)(p + 16);
  return u.v;
}

// ---------------- elementwise converters ----------------
__global__ void f2bf_kernel(const float* __restrict__ s, __bf16* __restrict__ d, int n) {
  for (int i = blockIdx.x * blockDim.x + threadIdx.x; i < n; i += gridDim.x * blockDim.x)
    d[i] = (__bf16)s[i];
}

// Wq [NH, H, DH] f32  ->  Wp [H, NH*DH] bf16  (head-concat columns)
__global__ void packW_kernel(const float* __restrict__ s, __bf16* __restrict__ d) {
  int n = CNH * CH * CDH;
  for (int i = blockIdx.x * blockDim.x + threadIdx.x; i < n; i += gridDim.x * blockDim.x) {
    int nh = i / (CH * CDH);
    int r  = i % (CH * CDH);
    int h  = r / CDH;
    int dd = r % CDH;
    d[(size_t)h * (CNH * CDH) + nh * CDH + dd] = (__bf16)s[i];
  }
}

// ---------------- generic bf16 GEMM:  C[M,N] = A[M,K] @ B[K,N] + bias ----------------
// epi: 0 = none, 1 = gelu(tanh), 2 = zero rows where token is padding
__global__ __launch_bounds__(256)
void gemm_bf16_kernel(const __bf16* __restrict__ A, const __bf16* __restrict__ Bm,
                      const float* __restrict__ bias,
                      float* __restrict__ outF, __bf16* __restrict__ outH,
                      int N, int K, int epi, const int* __restrict__ lengths) {
  __shared__ __bf16 lsA[64 * 32];     // A tile  [64 rows][32 k]
  __shared__ __bf16 lsBT[128 * 32];   // B tile transposed [128 n][32 k]

  const int tid  = threadIdx.x;
  const int lane = tid & 31;
  const int wave = tid >> 5;
  const int half = lane >> 4;
  const int ln16 = lane & 15;
  const int row0 = blockIdx.y * 64;
  const int col0 = blockIdx.x * 128;
  const int rt = wave & 3;            // row tile (16 rows)
  const int cg = (wave >> 2) * 4;     // col tile group base (4 tiles of 16)

  v8f acc[4];
#pragma unroll
  for (int i = 0; i < 4; i++)
#pragma unroll
    for (int r = 0; r < 8; r++) acc[i][r] = 0.0f;

  const int arow = tid >> 2;          // 0..63
  const int achk = (tid & 3) * 8;     // 0,8,16,24
  const int bk   = tid >> 4;          // 0..15
  const int bnc  = (tid & 15) * 8;    // 0..120

  for (int k0 = 0; k0 < K; k0 += 32) {
    __syncthreads();
    // stage A 64x32 (one b128 per thread)
    *(v8bf*)&lsA[arow * 32 + achk] =
        *(const v8bf*)&A[(size_t)(row0 + arow) * K + k0 + achk];
    // stage B transposed: coalesced b128 reads, scattered b16 LDS writes
#pragma unroll
    for (int kk = 0; kk < 32; kk += 16) {
      v8bf b8 = *(const v8bf*)&Bm[(size_t)(k0 + kk + bk) * N + col0 + bnc];
#pragma unroll
      for (int j = 0; j < 8; j++) lsBT[(bnc + j) * 32 + kk + bk] = b8[j];
    }
    __syncthreads();

    v16bf af = load_frag(&lsA[(rt * 16 + ln16) * 32 + half * 8]);
#pragma unroll
    for (int ct = 0; ct < 4; ct++) {
      v16bf bfg = load_frag(&lsBT[((cg + ct) * 16 + ln16) * 32 + half * 8]);
      acc[ct] = WMMA_BF16(af, bfg, acc[ct]);
    }
  }

  // epilogue: C/D layout -> (m = r + half*8, n = lane%16)
#pragma unroll
  for (int ct = 0; ct < 4; ct++) {
#pragma unroll
    for (int r = 0; r < 8; r++) {
      int row = row0 + rt * 16 + r + half * 8;
      int col = col0 + (cg + ct) * 16 + ln16;
      float v = acc[ct][r] + (bias ? bias[col] : 0.0f);
      if (epi == 1) {
        float x = v;
        v = 0.5f * x * (1.0f + tanhf(0.7978845608028654f * (x + 0.044715f * x * x * x)));
      } else if (epi == 2) {
        int bb = row >> 10, l = row & 1023;   // L = 1024
        if (l >= lengths[bb]) v = 0.0f;
      }
      if (outF) outF[(size_t)row * N + col] = v;
      if (outH) outH[(size_t)row * N + col] = (__bf16)v;
    }
  }
}

// ---------------- flash attention: per block (b, head, 128 queries) ----------------
__global__ __launch_bounds__(256)
void attn_flash_kernel(const __bf16* __restrict__ Q, const __bf16* __restrict__ Km,
                       const __bf16* __restrict__ V, __bf16* __restrict__ O) {
  __shared__ __bf16 lsK[32 * 64];     // K tile rows [32 keys][64 d]
  __shared__ __bf16 lsVT[64 * 32];    // V tile transposed [64 d][32 keys]
  __shared__ __bf16 lsP[8][16 * 32];  // per-wave probs [16 q][32 keys]

  const int tid  = threadIdx.x;
  const int lane = tid & 31;
  const int wave = tid >> 5;
  const int half = lane >> 4;
  const int ln16 = lane & 15;

  const int bid  = blockIdx.x;          // B*NH*(L/128) = 1024
  const int b    = bid >> 6;
  const int head = (bid >> 3) & 7;
  const int qblk = bid & 7;

  // Q fragments for this wave's 16 query rows (persist across key loop)
  const int qrow = qblk * 128 + wave * 16 + ln16;
  const __bf16* qp = Q + (size_t)(b * CL + qrow) * CH + head * CDH + half * 8;
  v16bf aq0 = load_frag(qp);        // d = 0..31
  v16bf aq1 = load_frag(qp + 32);   // d = 32..63

  float mrun[8], srun[8];
  v8f oacc[4];
#pragma unroll
  for (int r = 0; r < 8; r++) { mrun[r] = -3.0e38f; srun[r] = 0.0f; }
#pragma unroll
  for (int ct = 0; ct < 4; ct++)
#pragma unroll
    for (int r = 0; r < 8; r++) oacc[ct][r] = 0.0f;

  const float sc = 0.125f;            // 1/sqrt(DH)
  const int kr = tid >> 3;            // 0..31
  const int ch = (tid & 7) * 8;       // 0..56

  for (int kt = 0; kt < CL; kt += 32) {
    __syncthreads();   // prior iteration's V reads done before restage
    {
      size_t src = (size_t)(b * CL + kt + kr) * CH + head * CDH + ch;
      *(v8bf*)&lsK[kr * 64 + ch] = *(const v8bf*)&Km[src];
      v8bf v8 = *(const v8bf*)&V[src];
#pragma unroll
      for (int j = 0; j < 8; j++) lsVT[(ch + j) * 32 + kr] = v8[j];
    }
    __syncthreads();

    // S tiles: keys kt..kt+15 (s0) and kt+16..kt+31 (s1)
    v8f s0, s1;
#pragma unroll
    for (int r = 0; r < 8; r++) { s0[r] = 0.0f; s1[r] = 0.0f; }
    {
      v16bf bk0a = load_frag(&lsK[ln16 * 64 + half * 8]);
      v16bf bk0b = load_frag(&lsK[ln16 * 64 + 32 + half * 8]);
      s0 = WMMA_BF16(aq0, bk0a, s0);
      s0 = WMMA_BF16(aq1, bk0b, s0);
      v16bf bk1a = load_frag(&lsK[(16 + ln16) * 64 + half * 8]);
      v16bf bk1b = load_frag(&lsK[(16 + ln16) * 64 + 32 + half * 8]);
      s1 = WMMA_BF16(aq0, bk1a, s1);
      s1 = WMMA_BF16(aq1, bk1b, s1);
    }

    // online softmax per row (rows live in 16-lane groups; reduce with shfl_xor)
#pragma unroll
    for (int r = 0; r < 8; r++) {
      float v0 = s0[r] * sc, v1 = s1[r] * sc;
      float mx = fmaxf(v0, v1);
      mx = fmaxf(mx, __shfl_xor(mx, 1));
      mx = fmaxf(mx, __shfl_xor(mx, 2));
      mx = fmaxf(mx, __shfl_xor(mx, 4));
      mx = fmaxf(mx, __shfl_xor(mx, 8));
      float mnew = fmaxf(mrun[r], mx);
      float corr = __expf(mrun[r] - mnew);
      mrun[r] = mnew;
      float p0 = __expf(v0 - mnew);
      float p1 = __expf(v1 - mnew);
      float ps = p0 + p1;
      ps += __shfl_xor(ps, 1);
      ps += __shfl_xor(ps, 2);
      ps += __shfl_xor(ps, 4);
      ps += __shfl_xor(ps, 8);
      srun[r] = srun[r] * corr + ps;
#pragma unroll
      for (int ct = 0; ct < 4; ct++) oacc[ct][r] *= corr;
      lsP[wave][(r + half * 8) * 32 + ln16]      = (__bf16)p0;
      lsP[wave][(r + half * 8) * 32 + 16 + ln16] = (__bf16)p1;
    }
    __syncthreads();   // P visible; K tile reads done

    // O += P @ V  (contraction over 32 keys)
    v16bf ap = load_frag(&lsP[wave][ln16 * 32 + half * 8]);
#pragma unroll
    for (int ct = 0; ct < 4; ct++) {
      v16bf bv = load_frag(&lsVT[(ct * 16 + ln16) * 32 + half * 8]);
      oacc[ct] = WMMA_BF16(ap, bv, oacc[ct]);
    }
  }

  // normalize and store bf16 into [B*L, NH*DH] (head-concat)
#pragma unroll
  for (int ct = 0; ct < 4; ct++) {
#pragma unroll
    for (int r = 0; r < 8; r++) {
      int m = r + half * 8;
      int q = qblk * 128 + wave * 16 + m;
      float v = oacc[ct][r] / srun[r];
      O[(size_t)(b * CL + q) * CH + head * CDH + ct * 16 + ln16] = (__bf16)v;
    }
  }
}

// ---------------- fused residual + LayerNorm (+optional valid mask) ----------------
__global__ __launch_bounds__(256)
void ln_fuse_kernel(const float* __restrict__ X, const float* __restrict__ Yv,
                    const float* __restrict__ g, const float* __restrict__ bb,
                    float* __restrict__ outF, __bf16* __restrict__ outH,
                    const int* __restrict__ lengths) {
  __shared__ float r1[256], r2[256];
  const int row = blockIdx.x, tid = threadIdx.x;
  size_t ro = (size_t)row * CH;
  float x0 = X[ro + tid]       + Yv[ro + tid];
  float x1 = X[ro + tid + 256] + Yv[ro + tid + 256];
  r1[tid] = x0 + x1;
  r2[tid] = x0 * x0 + x1 * x1;
  __syncthreads();
  for (int s = 128; s > 0; s >>= 1) {
    if (tid < s) { r1[tid] += r1[tid + s]; r2[tid] += r2[tid + s]; }
    __syncthreads();
  }
  float mu  = r1[0] * (1.0f / (float)CH);
  float var = r2[0] * (1.0f / (float)CH) - mu * mu;
  float rs  = rsqrtf(var + 1e-5f);
  float vm  = 1.0f;
  if (lengths) { int b = row >> 10, l = row & 1023; vm = (l < lengths[b]) ? 1.0f : 0.0f; }
  float y0 = ((x0 - mu) * rs * g[tid]       + bb[tid])       * vm;
  float y1 = ((x1 - mu) * rs * g[tid + 256] + bb[tid + 256]) * vm;
  outF[ro + tid] = y0;
  outF[ro + tid + 256] = y1;
  if (outH) { outH[ro + tid] = (__bf16)y0; outH[ro + tid + 256] = (__bf16)y1; }
}

// ---------------------------------------------------------------------------------
extern "C" void kernel_launch(void* const* d_in, const int* in_sizes, int n_in,
                              void* d_out, int out_size, void* d_ws, size_t ws_size,
                              hipStream_t stream) {
  const float* values = (const float*)d_in[0];
  const int*   lengths = (const int*)d_in[1];
  const float* Wq = (const float*)d_in[2];
  const float* bq = (const float*)d_in[3];
  const float* Wk = (const float*)d_in[4];
  const float* bk = (const float*)d_in[5];
  const float* Wv = (const float*)d_in[6];
  const float* bv = (const float*)d_in[7];
  const float* Wc = (const float*)d_in[8];
  const float* bc = (const float*)d_in[9];
  const float* W1 = (const float*)d_in[10];
  const float* b1 = (const float*)d_in[11];
  const float* W2 = (const float*)d_in[12];
  const float* b2 = (const float*)d_in[13];
  const float* ln_g = (const float*)d_in[14];
  const float* ln_b = (const float*)d_in[15];
  float* out = (float*)d_out;

  // workspace layout (bytes); total ~174 MB with reuse.
  // Note: Qb/Kb/Vb/Obf are contiguous (4 x 16 MB) and are dead after the Wc
  // projection, so that 64 MB region is reused as the FFN hidden buffer Hbf.
  char* w = (char*)d_ws;
  size_t off = 0;
  auto alloc = [&](size_t bytes) { char* p = w + off; off += (bytes + 255) & ~(size_t)255; return p; };
  __bf16* Xbf  = (__bf16*)alloc((size_t)NTOK * CH * 2);        // 16 MB
  __bf16* Wqp  = (__bf16*)alloc((size_t)CH * CH * 2);
  __bf16* Wkp  = (__bf16*)alloc((size_t)CH * CH * 2);
  __bf16* Wvp  = (__bf16*)alloc((size_t)CH * CH * 2);
  __bf16* Wcb  = (__bf16*)alloc((size_t)CH * CH * 2);
  __bf16* W1b  = (__bf16*)alloc((size_t)CH * CM * 2);
  __bf16* W2b  = (__bf16*)alloc((size_t)CM * CH * 2);
  __bf16* Qb   = (__bf16*)alloc((size_t)NTOK * CH * 2);        // 16 MB (Hbf part 1)
  __bf16* Kb   = (__bf16*)alloc((size_t)NTOK * CH * 2);        // 16 MB (Hbf part 2)
  __bf16* Vb   = (__bf16*)alloc((size_t)NTOK * CH * 2);        // 16 MB (Hbf part 3)
  __bf16* Obf  = (__bf16*)alloc((size_t)NTOK * CH * 2);        // 16 MB (Hbf part 4)
  float*  tmp1 = (float*)alloc((size_t)NTOK * CH * 4);         // 32 MB (also tmp2)
  float*  aF   = (float*)alloc((size_t)NTOK * CH * 4);         // 32 MB
  __bf16* aH   = (__bf16*)alloc((size_t)NTOK * CH * 2);        // 16 MB
  __bf16* Hbf  = Qb;   // alias dead Q/K/V/O region for FFN hidden [NTOK, M]
  (void)ws_size; (void)in_sizes; (void)n_in; (void)out_size;

  // 1) convert inputs / pack weights
  f2bf_kernel<<<4096, 256, 0, stream>>>(values, Xbf, NTOK * CH);
  packW_kernel<<<1024, 256, 0, stream>>>(Wq, Wqp);
  packW_kernel<<<1024, 256, 0, stream>>>(Wk, Wkp);
  packW_kernel<<<1024, 256, 0, stream>>>(Wv, Wvp);
  f2bf_kernel<<<1024, 256, 0, stream>>>(Wc, Wcb, CH * CH);
  f2bf_kernel<<<4096, 256, 0, stream>>>(W1, W1b, CH * CM);
  f2bf_kernel<<<4096, 256, 0, stream>>>(W2, W2b, CM * CH);

  // 2) QKV projections (+bias, zero padded rows)
  dim3 gH(CH / 128, NTOK / 64);
  gemm_bf16_kernel<<<gH, 256, 0, stream>>>(Xbf, Wqp, bq, nullptr, Qb, CH, CH, 2, lengths);
  gemm_bf16_kernel<<<gH, 256, 0, stream>>>(Xbf, Wkp, bk, nullptr, Kb, CH, CH, 2, lengths);
  gemm_bf16_kernel<<<gH, 256, 0, stream>>>(Xbf, Wvp, bv, nullptr, Vb, CH, CH, 2, lengths);

  // 3) flash attention -> Obf [NTOK, NH*DH]
  attn_flash_kernel<<<CB * CNH * (CL / 128), 256, 0, stream>>>(Qb, Kb, Vb, Obf);

  // 4) output projection, residual + LN -> aF (f32), aH (bf16)
  gemm_bf16_kernel<<<gH, 256, 0, stream>>>(Obf, Wcb, bc, tmp1, nullptr, CH, CH, 0, nullptr);
  ln_fuse_kernel<<<NTOK, 256, 0, stream>>>(values, tmp1, ln_g, ln_b, aF, aH, nullptr);

  // 5) FFN: gelu(a@W1+b1) -> Hbf ; Hbf@W2+b2 -> tmp1
  dim3 gM(CM / 128, NTOK / 64);
  gemm_bf16_kernel<<<gM, 256, 0, stream>>>(aH, W1b, b1, nullptr, Hbf, CM, CH, 1, nullptr);
  gemm_bf16_kernel<<<gH, 256, 0, stream>>>(Hbf, W2b, b2, tmp1, nullptr, CH, CM, 0, nullptr);

  // 6) final residual + LN, mask padded rows -> out (f32)
  ln_fuse_kernel<<<NTOK, 256, 0, stream>>>(aF, tmp1, ln_g, ln_b, out, nullptr, lengths);
}
